// BloodFlowSkinAnalyzer_2542620639475
// MI455X (gfx1250) — compile-verified, matrix operand order checked by reading, not run on hardware
//
#include <hip/hip_runtime.h>
#include <hip/hip_bf16.h>
#include <math.h>

typedef __bf16 bf16_t;
typedef __attribute__((ext_vector_type(8)))  __bf16 v8bf;
typedef __attribute__((ext_vector_type(16))) __bf16 v16bf;
typedef __attribute__((ext_vector_type(8)))  float  v8f;
typedef __attribute__((ext_vector_type(4)))  int    v4i;

#define AS_GLOBAL __attribute__((address_space(1)))
#define AS_LOCAL  __attribute__((address_space(3)))

#if defined(__has_builtin)
#if __has_builtin(__builtin_amdgcn_global_load_async_to_lds_b128)
#define HAVE_ASYNC_B128 1
#endif
#endif

// Padded activation slab geometry (1-pixel zero halo):
//   h1[nl][py:130][px:130][ci:32] bf16, per-image stride 130*130*32 = 540800
#define PAD_IMG_ELEMS 540800
#define PAD_ROW_ELEMS (130 * 32)

// ---------------------------------------------------------------------------
// Kernel 0: prepack conv2 weights [16co,32ci,3,3] (f32) into per-lane WMMA
// B-fragments: wpack[tap][lane][slot] (bf16), 9*32*16 = 4608 elems.
// Lane l: co=l&15, hi=l>>4; slot s -> ci = (s<8) ? 8*hi+s : 16+8*hi+(s-8)
// ---------------------------------------------------------------------------
__global__ __launch_bounds__(288)
void wpack_kernel(const float* __restrict__ w2, bf16_t* __restrict__ wpack) {
    int t = threadIdx.x;                 // 0..287 = tap*32 + lane
    if (t >= 288) return;
    int tap  = t >> 5;
    int lane = t & 31;
    int co = lane & 15, hi = lane >> 4;
    #pragma unroll
    for (int s = 0; s < 16; ++s) {
        int ci = (s < 8) ? (8 * hi + s) : (16 + 8 * hi + (s - 8));
        wpack[t * 16 + s] = (bf16_t)w2[co * 288 + ci * 9 + tap];   // OIHW
    }
}

// ---------------------------------------------------------------------------
// Kernel 0b: zero the padded slab once per launch (halo stays zero; conv1
// rewrites the interior every chunk).
// ---------------------------------------------------------------------------
__global__ __launch_bounds__(256)
void zero_kernel(ulonglong2* __restrict__ p, long n16) {
    long i = (long)blockIdx.x * 256 + threadIdx.x;
    if (i < n16) { ulonglong2 z; z.x = 0; z.y = 0; p[i] = z; }
}

// ---------------------------------------------------------------------------
// Kernel 1: conv1 3->32 k3 p1, ReLU. Direct VALU, weights scalarized to
// s_loads (wave-uniform). Output: padded NHWC bf16 slab interior.
// ---------------------------------------------------------------------------
__global__ __launch_bounds__(256)
void conv1_kernel(const float* __restrict__ frames,   // [256,3,128,128]
                  const float* __restrict__ w1,       // [32,3,3,3]
                  const float* __restrict__ b1,       // [32]
                  bf16_t* __restrict__ h1,            // padded slab
                  int n0) {
    int pix = blockIdx.x * 256 + threadIdx.x;   // 0 .. 32*16384-1 (chunk-local)
    int nl = pix >> 14;
    int p  = pix & 16383;
    int hh = p >> 7, ww = p & 127;
    int n  = n0 + nl;

    float in[27];
    #pragma unroll
    for (int c = 0; c < 3; ++c) {
        const float* f = frames + ((size_t)(n * 3 + c) << 14);
        #pragma unroll
        for (int dy = 0; dy < 3; ++dy) {
            int y = hh + dy - 1;
            #pragma unroll
            for (int dx = 0; dx < 3; ++dx) {
                int x = ww + dx - 1;
                in[c * 9 + dy * 3 + dx] =
                    ((unsigned)y < 128u && (unsigned)x < 128u) ? f[y * 128 + x] : 0.f;
            }
        }
    }
    union { bf16_t e[32]; v8bf v[4]; } outv;
    for (int co = 0; co < 32; ++co) {           // rolled: 27 s_loads/iter
        float a = b1[co];
        const float* w = w1 + co * 27;
        #pragma unroll
        for (int k = 0; k < 27; ++k) a = fmaf(w[k], in[k], a);
        a = a > 0.f ? a : 0.f;
        outv.e[co] = (bf16_t)a;
    }
    bf16_t* dst = h1 + ((size_t)nl * 16900 + (size_t)(hh + 1) * 130 + (ww + 1)) * 32;
    *(v8bf*)(dst +  0) = outv.v[0];
    *(v8bf*)(dst +  8) = outv.v[1];
    *(v8bf*)(dst + 16) = outv.v[2];
    *(v8bf*)(dst + 24) = outv.v[3];
}

// ---------------------------------------------------------------------------
// Kernel 2: conv2 32->16 k3 p1 + ReLU, fused conv3 16->1 k1 + sigmoid.
// One wave = 16-pixel x 16-channel tile, 9x v_wmma_f32_16x16x32_bf16.
// Halo-padded input => branchless A loads with immediate tap offsets.
// B-fragments async-staged to LDS once per block, read via ds_load.
// ---------------------------------------------------------------------------
__global__ __launch_bounds__(256)
void conv2_wmma_kernel(const bf16_t* __restrict__ h1,     // padded NHWC slab
                       const bf16_t* __restrict__ wpack,  // [9][32][16]
                       const float*  __restrict__ b2,     // [16]
                       const float*  __restrict__ w3,     // [16]
                       const float*  __restrict__ b3p,    // [1]
                       float* __restrict__ masks_out,     // [256,128,128]
                       int n0) {
    __shared__ __align__(32) bf16_t sB[4608];              // 9216 B

    // Stage B-fragments into LDS (async DMA path when available).
#ifdef HAVE_ASYNC_B128
    for (int i = threadIdx.x; i < 576; i += 256) {
        const void* gsrc = (const void*)(wpack + (i << 3));
        void*       ldst = (void*)(sB + (i << 3));
        __builtin_amdgcn_global_load_async_to_lds_b128(
            (AS_GLOBAL v4i*)(AS_GLOBAL void*)const_cast<void*>(gsrc),
            (AS_LOCAL  v4i*)(AS_LOCAL  void*)ldst, 0, 0);
    }
#if __has_builtin(__builtin_amdgcn_s_wait_asynccnt)
    __builtin_amdgcn_s_wait_asynccnt(0);
#else
    asm volatile("s_wait_asynccnt 0x0" ::: "memory");
#endif
#else
    for (int i = threadIdx.x; i < 4608; i += 256) sB[i] = wpack[i];
#endif
    __syncthreads();

    int wave = threadIdx.x >> 5;
    int lane = threadIdx.x & 31;
    int tile = blockIdx.x * 8 + wave;     // 0 .. 32*1024-1
    int nl  = tile >> 10;
    int rem = tile & 1023;
    int h   = rem >> 3;
    int w0  = (rem & 7) << 4;
    int m = lane & 15, hi = lane >> 4;

    // Per-lane base at padded (row h, col w0+m) == tap (kh=0,kw=0).
    const bf16_t* base = h1 + ((size_t)nl * 16900 + (size_t)h * 130 + (w0 + m)) * 32
                            + 8 * hi;
    v8f acc = {};
    #pragma unroll
    for (int kh = 0; kh < 3; ++kh) {
        #pragma unroll
        for (int kw = 0; kw < 3; ++kw) {
            const bf16_t* p = base + (kh * PAD_ROW_ELEMS + kw * 32);
            v8bf a0 = *(const v8bf*)(p);        // ci = 8*hi .. 8*hi+7
            v8bf a1 = *(const v8bf*)(p + 16);   // ci = 16+8*hi ..
            v16bf Av = __builtin_shufflevector(a0, a1,
                         0, 1, 2, 3, 4, 5, 6, 7, 8, 9, 10, 11, 12, 13, 14, 15);
            int tap = kh * 3 + kw;
            v16bf Bv = *(const v16bf*)(sB + ((tap * 32 + lane) << 4));
            acc = __builtin_amdgcn_wmma_f32_16x16x32_bf16(
                      false, Av, false, Bv, (short)0, acc, false, false);
        }
    }

    // Epilogue: val = relu(acc + b2[co]); mask = sigmoid(sum_co val*w3[co] + b3)
    float bias = b2[m];
    float cw3  = w3[m];
    float b3   = b3p[0];
    int n = n0 + nl;
    float* mrow = masks_out + ((size_t)n << 14) + h * 128 + w0;
    #pragma unroll
    for (int j = 0; j < 8; ++j) {
        float val = acc[j] + bias;
        val = val > 0.f ? val : 0.f;
        float s = val * cw3;
        s += __shfl_xor(s, 1, 32);
        s += __shfl_xor(s, 2, 32);
        s += __shfl_xor(s, 4, 32);
        s += __shfl_xor(s, 8, 32);          // full sum within each 16-lane half
        float mk = 1.f / (1.f + __expf(-(s + b3)));
        if (m == 0) mrow[j + 8 * hi] = mk;  // pixel row M = j + 8*hi
    }
}

// ---------------------------------------------------------------------------
// Kernel 3: masked per-channel mean -> color_signals [256,3]
// ---------------------------------------------------------------------------
__global__ __launch_bounds__(256)
void colorsig_kernel(const float* __restrict__ frames,
                     const float* __restrict__ masks,
                     float* __restrict__ colors) {
    int n = blockIdx.x;
    const float* mk = masks  + ((size_t)n << 14);
    const float* f0 = frames + ((size_t)(n * 3) << 14);
    float sm = 0, s0 = 0, s1 = 0, s2 = 0;
    for (int p = threadIdx.x; p < 16384; p += 256) {
        float mv = mk[p];
        sm += mv;
        s0 = fmaf(mv, f0[p],         s0);
        s1 = fmaf(mv, f0[p + 16384], s1);
        s2 = fmaf(mv, f0[p + 32768], s2);
    }
    __shared__ float r[4][256];
    r[0][threadIdx.x] = sm; r[1][threadIdx.x] = s0;
    r[2][threadIdx.x] = s1; r[3][threadIdx.x] = s2;
    __syncthreads();
    for (int off = 128; off > 0; off >>= 1) {
        if ((int)threadIdx.x < off)
            for (int k = 0; k < 4; ++k)
                r[k][threadIdx.x] += r[k][threadIdx.x + off];
        __syncthreads();
    }
    if (threadIdx.x == 0) {
        float denom = r[0][0] + 1e-8f;
        colors[n * 3 + 0] = r[1][0] / denom;
        colors[n * 3 + 1] = r[2][0] / denom;
        colors[n * 3 + 2] = r[3][0] / denom;
    }
}

// ---------------------------------------------------------------------------
// Kernel 4: entire temporal tail (conv1d chains, FC, std/var) — one block.
// ---------------------------------------------------------------------------
__global__ __launch_bounds__(64)
void tail_kernel(const float* __restrict__ colors,
                 const float* ct_w1, const float* ct_b1,
                 const float* ct_w2, const float* ct_b2,
                 const float* ct_w3, const float* ct_b3,
                 const float* ta_w1, const float* ta_b1,
                 const float* ta_w2, const float* ta_b2,
                 const float* ta_w3, const float* ta_b3,
                 const float* fc_w1, const float* fc_b1,
                 const float* fc_w2, const float* fc_b2,
                 const float* fc_w3, const float* fc_b3,
                 float* __restrict__ flow, float* __restrict__ syncsc,
                 float* __restrict__ nat,  float* __restrict__ tf) {
    __shared__ float CS[3][64];
    __shared__ float E[3][64];
    __shared__ float F1[32][64];
    __shared__ float F2[64][64];
    __shared__ float F3[32][64];
    __shared__ float P[32];
    __shared__ float G1[64];
    __shared__ float G2[32];
    __shared__ float SD[3];
    int t = threadIdx.x;   // 0..63, one time-step per thread

    for (int b = 0; b < 4; ++b) {
        float c0 = colors[(b * 64 + t) * 3 + 0];
        float c1 = colors[(b * 64 + t) * 3 + 1];
        float c2 = colors[(b * 64 + t) * 3 + 2];
        CS[0][t] = c0; CS[1][t] = c1; CS[2][t] = c2;

        // color_transformer (k=1): 3->16->8->3 (last layer no relu)
        float h16[16];
        #pragma unroll
        for (int o = 0; o < 16; ++o) {
            float a = ct_b1[o] + ct_w1[o*3+0]*c0 + ct_w1[o*3+1]*c1 + ct_w1[o*3+2]*c2;
            h16[o] = a > 0.f ? a : 0.f;
        }
        float h8[8];
        #pragma unroll
        for (int o = 0; o < 8; ++o) {
            float a = ct_b2[o];
            #pragma unroll
            for (int i = 0; i < 16; ++i) a = fmaf(ct_w2[o*16+i], h16[i], a);
            h8[o] = a > 0.f ? a : 0.f;
        }
        #pragma unroll
        for (int c = 0; c < 3; ++c) {
            float a = ct_b3[c];
            #pragma unroll
            for (int i = 0; i < 8; ++i) a = fmaf(ct_w3[c*8+i], h8[i], a);
            E[c][t] = a;
        }
        __syncthreads();

        // temporal_analyzer conv1: 3->32 k3 p1 relu
        for (int co = 0; co < 32; ++co) {
            float a = ta_b1[co];
            #pragma unroll
            for (int ci = 0; ci < 3; ++ci) {
                const float* w = ta_w1 + co*9 + ci*3;
                if (t > 0)  a = fmaf(w[0], E[ci][t-1], a);
                a = fmaf(w[1], E[ci][t], a);
                if (t < 63) a = fmaf(w[2], E[ci][t+1], a);
            }
            F1[co][t] = a > 0.f ? a : 0.f;
        }
        __syncthreads();
        // conv2: 32->64 k3 p1 relu
        for (int co = 0; co < 64; ++co) {
            float a = ta_b2[co];
            for (int ci = 0; ci < 32; ++ci) {
                const float* w = ta_w2 + co*96 + ci*3;
                if (t > 0)  a = fmaf(w[0], F1[ci][t-1], a);
                a = fmaf(w[1], F1[ci][t], a);
                if (t < 63) a = fmaf(w[2], F1[ci][t+1], a);
            }
            F2[co][t] = a > 0.f ? a : 0.f;
        }
        __syncthreads();
        // conv3: 64->32 k3 p1 relu, also stream out temporal_features
        for (int co = 0; co < 32; ++co) {
            float a = ta_b3[co];
            for (int ci = 0; ci < 64; ++ci) {
                const float* w = ta_w3 + co*192 + ci*3;
                if (t > 0)  a = fmaf(w[0], F2[ci][t-1], a);
                a = fmaf(w[1], F2[ci][t], a);
                if (t < 63) a = fmaf(w[2], F2[ci][t+1], a);
            }
            float v = a > 0.f ? a : 0.f;
            F3[co][t] = v;
            tf[b * 2048 + co * 64 + t] = v;
        }
        __syncthreads();

        if (t < 32) {                         // mean-pool over T
            float a = 0.f;
            for (int k = 0; k < 64; ++k) a += F3[t][k];
            P[t] = a * (1.f / 64.f);
        }
        __syncthreads();
        {                                     // FC 32->64 relu
            float a = fc_b1[t];
            #pragma unroll
            for (int i = 0; i < 32; ++i) a = fmaf(fc_w1[t*32+i], P[i], a);
            G1[t] = a > 0.f ? a : 0.f;
        }
        __syncthreads();
        if (t < 32) {                         // FC 64->32 relu
            float a = fc_b2[t];
            #pragma unroll
            for (int i = 0; i < 64; ++i) a = fmaf(fc_w2[t*64+i], G1[i], a);
            G2[t] = a > 0.f ? a : 0.f;
        }
        __syncthreads();
        if (t < 3) {                          // per-channel std (ddof=1)
            float mean = 0.f;
            for (int k = 0; k < 64; ++k) mean += CS[t][k];
            mean *= (1.f / 64.f);
            float var = 0.f;
            for (int k = 0; k < 64; ++k) { float d = CS[t][k] - mean; var = fmaf(d, d, var); }
            SD[t] = sqrtf(var * (1.f / 63.f));
        }
        __syncthreads();
        if (t == 0) {
            float s = fc_b3[0];
            #pragma unroll
            for (int i = 0; i < 32; ++i) s = fmaf(fc_w3[i], G2[i], s);
            float fl = 1.f / (1.f + __expf(-s));
            float mm = (SD[0] + SD[1] + SD[2]) * (1.f / 3.f);
            float d0 = SD[0]-mm, d1 = SD[1]-mm, d2 = SD[2]-mm;
            float v2 = (d0*d0 + d1*d1 + d2*d2) * 0.5f;   // var ddof=1 over 3
            float sy = 1.f / (1.f + __expf(v2));         // sigmoid(-v2)
            flow[b]   = fl;
            syncsc[b] = sy;
            nat[b]    = 0.5f * (fl + sy);
        }
        __syncthreads();
    }
}

// ---------------------------------------------------------------------------
// Launcher. d_out layout (all f32, concatenated in return order):
//   [0..3] flow, [4..7] sync, [8..775] color_signals(256*3),
//   [776..4195079] skin_masks(256*16384), [4195080..83] naturalness,
//   [4195084..4203275] temporal_features(4*32*64)
// Workspace: wpack (16 KB slot) + padded h1 bf16 slab for a 32-image chunk
//   (32 * 130*130*32 * 2 B = 34,611,200 B)
// ---------------------------------------------------------------------------
extern "C" void kernel_launch(void* const* d_in, const int* in_sizes, int n_in,
                              void* d_out, int out_size, void* d_ws, size_t ws_size,
                              hipStream_t stream) {
    const float* frames = (const float*)d_in[0];
    const float* seg_w1 = (const float*)d_in[1];
    const float* seg_b1 = (const float*)d_in[2];
    const float* seg_w2 = (const float*)d_in[3];
    const float* seg_b2 = (const float*)d_in[4];
    const float* seg_w3 = (const float*)d_in[5];
    const float* seg_b3 = (const float*)d_in[6];
    const float* ct_w1 = (const float*)d_in[7];
    const float* ct_b1 = (const float*)d_in[8];
    const float* ct_w2 = (const float*)d_in[9];
    const float* ct_b2 = (const float*)d_in[10];
    const float* ct_w3 = (const float*)d_in[11];
    const float* ct_b3 = (const float*)d_in[12];
    const float* ta_w1 = (const float*)d_in[13];
    const float* ta_b1 = (const float*)d_in[14];
    const float* ta_w2 = (const float*)d_in[15];
    const float* ta_b2 = (const float*)d_in[16];
    const float* ta_w3 = (const float*)d_in[17];
    const float* ta_b3 = (const float*)d_in[18];
    const float* fc_w1 = (const float*)d_in[19];
    const float* fc_b1 = (const float*)d_in[20];
    const float* fc_w2 = (const float*)d_in[21];
    const float* fc_b2 = (const float*)d_in[22];
    const float* fc_w3 = (const float*)d_in[23];
    const float* fc_b3 = (const float*)d_in[24];

    float* out    = (float*)d_out;
    float* flow   = out + 0;
    float* syncsc = out + 4;
    float* colors = out + 8;
    float* masks  = out + 776;
    float* nat    = out + 4195080;
    float* tf     = out + 4195084;

    bf16_t* wpack = (bf16_t*)d_ws;
    bf16_t* h1    = (bf16_t*)((char*)d_ws + 16384);
    const long h1_bytes = 32L * PAD_IMG_ELEMS * 2;          // 34,611,200
    const long h1_u16   = h1_bytes / 16;                    // 2,163,200

    wpack_kernel<<<1, 288, 0, stream>>>(seg_w2, wpack);
    zero_kernel<<<(int)((h1_u16 + 255) / 256), 256, 0, stream>>>(
        (ulonglong2*)h1, h1_u16);

    // 8 chunks of 32 images; stream order serializes conv1 -> conv2 per chunk
    for (int c = 0; c < 8; ++c) {
        int n0 = c * 32;
        conv1_kernel<<<2048, 256, 0, stream>>>(frames, seg_w1, seg_b1, h1, n0);
        conv2_wmma_kernel<<<4096, 256, 0, stream>>>(h1, wpack, seg_b2, seg_w3,
                                                    seg_b3, masks, n0);
    }
    colorsig_kernel<<<256, 256, 0, stream>>>(frames, masks, colors);
    tail_kernel<<<1, 64, 0, stream>>>(colors,
        ct_w1, ct_b1, ct_w2, ct_b2, ct_w3, ct_b3,
        ta_w1, ta_b1, ta_w2, ta_b2, ta_w3, ta_b3,
        fc_w1, fc_b1, fc_w2, fc_b2, fc_w3, fc_b3,
        flow, syncsc, nat, tf);
}
